// MixBlock3D_66176856097154
// MI455X (gfx1250) — matrix operand order repaired and checked by reading, not compiled, not run on hardware
//
#include <hip/hip_runtime.h>
#include <hip/hip_bf16.h>

// ---------------------------------------------------------------------------
// MixBlock3D for MI455X (gfx1250): all GEMM-shaped work (attention qkv/scores/
// PV/proj + 3x3x3 convs as tap-wise implicit GEMM) via v_wmma_f32_16x16x32_f16.
// Compute-bound (~240 GFLOP vs ~10us of HBM time at 23.3 TB/s), so matrix
// cores are the whole game. wave32 everywhere. Dual accumulators break the
// WMMA D->C RAW chains (ISA 7.12.1); weights prefetched with
// global_prefetch_b8 (L2-resident, 192MB L2).
// ---------------------------------------------------------------------------

typedef __attribute__((ext_vector_type(16))) _Float16 v16h;
typedef __attribute__((ext_vector_type(8)))  float    v8f;

#define DEVINL __device__ __forceinline__

constexpr int NTOK = 98;     // tokens per window/grid group
constexpr int MPAD = 112;    // padded to 7 wmma M-tiles
constexpr int NWIN = 512;    // 2 * 4 * 8 * 8
constexpr int NPOS = 2 * 8 * 56 * 56; // 50176 voxels

union Frag {
    v16h v;
    _Float16 h[16];
    uint4 q[2];
};

// A-matrix 16x32 f16 fragment (ISA 7.12.2): lanes 0-15 -> row M=lane,
// halves K=[0..7],[16..23]; lanes 16-31 -> row M=lane-16, K=[8..15],[24..31].
DEVINL v16h load_frag_a(const _Float16* base, int ld) {
    int lane = threadIdx.x & 31;
    int r = lane & 15;
    int ko = (lane < 16) ? 0 : 8;
    const _Float16* p = base + r * ld + ko;
    Frag f;
    f.q[0] = *(const uint4*)(p);
    f.q[1] = *(const uint4*)(p + 16);
    return f.v;
}

// B-matrix 32x16 f16 fragment: lanes 0-15 -> col N=lane, K=[0..7],[16..23];
// lanes 16-31 -> col N=lane-16, K=[8..15],[24..31].  Storage is row-major with rows =
// output column (weight row / token), i.e. K contiguous along a row; each lane reads two 8-K runs.
DEVINL v16h load_frag_b(const _Float16* base, int ld) {
    int lane = threadIdx.x & 31;
    int c = lane & 15;
    int ko = (lane < 16) ? 0 : 8;
    const _Float16* p = base + c * ld + ko;
    Frag f;
    f.q[0] = *(const uint4*)(p);
    f.q[1] = *(const uint4*)(p + 16);
    return f.v;
}

// A fragment for conv rows with W-boundary zero guard.
DEVINL v16h load_frag_a_conv(const _Float16* row0, int wbase, int kofs) {
    int lane = threadIdx.x & 31;
    int r = lane & 15;
    int ko = (lane < 16) ? 0 : 8;
    int wv = wbase + r;
    Frag f;
    if (wv >= 0 && wv < 56) {
        const _Float16* p = row0 + wv * 128 + kofs + ko;
        f.q[0] = *(const uint4*)(p);
        f.q[1] = *(const uint4*)(p + 16);
    } else {
        f.q[0] = make_uint4(0u, 0u, 0u, 0u);
        f.q[1] = make_uint4(0u, 0u, 0u, 0u);
    }
    return f.v;
}

DEVINL v8f wmma_acc(v16h a, v16h b, v8f c) {
    return __builtin_amdgcn_wmma_f32_16x16x32_f16(false, a, false, b,
                                                  (short)0, c, false, false);
}

// ---------------------------------------------------------------------------
// Weight conversion
// ---------------------------------------------------------------------------
__global__ void cvt_f16_kernel(const float* __restrict__ src,
                               _Float16* __restrict__ dst, int n) {
    int i = blockIdx.x * blockDim.x + threadIdx.x;
    if (i < n) dst[i] = (_Float16)src[i];
}

// (oc,ic,3,3,3) f32 -> tap-major [27][oc=128][ic=128] f16
__global__ void cvt_convw_kernel(const float* __restrict__ src,
                                 _Float16* __restrict__ dst) {
    int i = blockIdx.x * blockDim.x + threadIdx.x;
    if (i < 27 * 128 * 128) {
        int tap = i >> 14;
        int r2 = i & 16383;
        int oc = r2 >> 7;
        int ic = r2 & 127;
        dst[i] = (_Float16)src[(size_t)(oc * 128 + ic) * 27 + tap];
    }
}

// ---------------------------------------------------------------------------
// LayerNorm + window/grid gather -> f16 token matrix [512][112][256] (pad=0)
// gridMode==0: src = input (B,C,D,H,W) f32. gridMode==1: src = XW channel-last.
// ---------------------------------------------------------------------------
__global__ __launch_bounds__(256)
void ln_gather_kernel(const float* __restrict__ src,
                      const float* __restrict__ gw,
                      const float* __restrict__ gb,
                      _Float16* __restrict__ dst, int gridMode) {
    int wi = blockIdx.x;
    int b = wi >> 8;
    int rem = wi & 255;
    int gd = rem >> 6, gh = (rem >> 3) & 7, gwn = rem & 7;
    int wave = threadIdx.x >> 5, lane = threadIdx.x & 31;

    for (int t = wave; t < MPAD; t += 8) {
        _Float16* outp = dst + ((size_t)wi * MPAD + t) * 256;
        if (t >= NTOK) {
            for (int j = 0; j < 8; ++j) outp[j * 32 + lane] = (_Float16)0.f;
            continue;
        }
        int dz = t / 49, hy = (t % 49) / 7, wx = t % 7;
        int dv, hv, wv;
        if (gridMode) { dv = dz * 4 + gd; hv = hy * 8 + gh; wv = wx * 8 + gwn; }
        else          { dv = gd * 2 + dz; hv = gh * 7 + hy; wv = gwn * 7 + wx; }

        float vals[8];
        float s = 0.f;
        if (gridMode) {
            const float* p = src + ((size_t)((b * 8 + dv) * 56 + hv) * 56 + wv) * 256;
            for (int j = 0; j < 8; ++j) { vals[j] = p[j * 32 + lane]; s += vals[j]; }
        } else {
            size_t sp = (size_t)hv * 56 + wv;
            for (int j = 0; j < 8; ++j) {
                int c = j * 32 + lane;
                vals[j] = src[((size_t)(b * 256 + c) * 8 + dv) * 3136 + sp];
                s += vals[j];
            }
        }
        for (int off = 16; off; off >>= 1) s += __shfl_xor(s, off, 32);
        float mean = s * (1.f / 256.f);
        float sq = 0.f;
        for (int j = 0; j < 8; ++j) { float d = vals[j] - mean; sq += d * d; }
        for (int off = 16; off; off >>= 1) sq += __shfl_xor(sq, off, 32);
        float rinv = rsqrtf(sq * (1.f / 256.f) + 1e-5f);
        for (int j = 0; j < 8; ++j) {
            int c = j * 32 + lane;
            outp[c] = (_Float16)((vals[j] - mean) * rinv * gw[c] + gb[c]);
        }
    }
}

// ---------------------------------------------------------------------------
// QKV GEMM: per window, (112x256) x (256x768) -> Q(scaled), K, V^T (f16)
// q,k: [512][4][112][64]   vt: [512][4][64][128] (cols 112..127 zeroed)
// ---------------------------------------------------------------------------
__global__ __launch_bounds__(256)
void qkv_gemm_kernel(const _Float16* __restrict__ X,
                     const _Float16* __restrict__ Wh,
                     _Float16* __restrict__ qb,
                     _Float16* __restrict__ kb,
                     _Float16* __restrict__ vt) {
    __shared__ _Float16 Xs[MPAD * 264];
    int wi = blockIdx.x;
    int tid = threadIdx.x;

    // zero V^T pad columns so PV GEMM can use K=128
    for (int idx = tid; idx < 4 * 64 * 16; idx += 256) {
        int head = idx >> 10;
        int r2 = idx & 1023;
        int dim = r2 >> 4;
        int cp = 112 + (r2 & 15);
        vt[(((size_t)wi * 4 + head) * 64 + dim) * 128 + cp] = (_Float16)0.f;
    }
    const _Float16* Xg = X + (size_t)wi * MPAD * 256;
    for (int idx = tid; idx < MPAD * 32; idx += 256) {
        int row = idx >> 5, cv = idx & 31;
        *(uint4*)(&Xs[row * 264 + cv * 8]) = *(const uint4*)(Xg + row * 256 + cv * 8);
    }
    __syncthreads();

    int wave = tid >> 5, lane = tid & 31;
    for (int tile = wave; tile < 7 * 48; tile += 8) {
        int mt = tile / 48, nt = tile % 48;
        const _Float16* wrow = Wh + (size_t)(nt * 16) * 256;
        __builtin_prefetch(wrow, 0, 1);              // global_prefetch_b8
        v8f acc0 = {}, acc1 = {};
        for (int kk = 0; kk < 8; kk += 2) {
            v16h a0 = load_frag_a(&Xs[mt * 16 * 264 + kk * 32], 264);
            v16h b0 = load_frag_b(wrow + kk * 32, 256);
            acc0 = wmma_acc(a0, b0, acc0);
            v16h a1 = load_frag_a(&Xs[mt * 16 * 264 + (kk + 1) * 32], 264);
            v16h b1 = load_frag_b(wrow + (kk + 1) * 32, 256);
            acc1 = wmma_acc(a1, b1, acc1);
        }
        v8f acc = acc0 + acc1;
        int n = nt * 16 + (lane & 15);
        int region = n >> 8;          // 0=q 1=k 2=v
        int jj = n & 255;
        int head = jj >> 6, dim = jj & 63;
        int mofs = (lane < 16) ? 0 : 8;
        for (int i = 0; i < 8; ++i) {
            int m = mt * 16 + i + mofs;
            float v = acc[i];
            if (region == 0)
                qb[(((size_t)wi * 4 + head) * 112 + m) * 64 + dim] = (_Float16)(v * 0.125f);
            else if (region == 1)
                kb[(((size_t)wi * 4 + head) * 112 + m) * 64 + dim] = (_Float16)v;
            else
                vt[(((size_t)wi * 4 + head) * 64 + dim) * 128 + m] = (_Float16)v;
        }
    }
}

// ---------------------------------------------------------------------------
// Attention core per (window, head): S = Q K^T (+bias) -> softmax -> P V
// Scores f32 in LDS; P f16 in its own LDS array (no type-punned overlap).
// AO: [512][112][256] f16 (heads concatenated).
// ---------------------------------------------------------------------------
__global__ __launch_bounds__(128)
void attn_core_kernel(const _Float16* __restrict__ qb,
                      const _Float16* __restrict__ kb,
                      const _Float16* __restrict__ vt,
                      const float* __restrict__ bt,
                      _Float16* __restrict__ AO) {
    __shared__ float Ssc[MPAD * 116];
    __shared__ _Float16 P[MPAD * 136];
    const int PLD = 136;

    int wi = blockIdx.x >> 2;
    int head = blockIdx.x & 3;
    int wave = threadIdx.x >> 5, lane = threadIdx.x & 31;

    const _Float16* qg = qb + (((size_t)wi * 4 + head) * 112) * 64;
    const _Float16* kg = kb + (((size_t)wi * 4 + head) * 112) * 64;
    const _Float16* vg = vt + (((size_t)wi * 4 + head) * 64) * 128;

    for (int tile = wave; tile < 49; tile += 4) {
        int mt = tile / 7, nt = tile % 7;
        v16h a0 = load_frag_a(qg + mt * 16 * 64, 64);
        v16h b0 = load_frag_b(kg + nt * 16 * 64, 64);
        v16h a1 = load_frag_a(qg + mt * 16 * 64 + 32, 64);
        v16h b1 = load_frag_b(kg + nt * 16 * 64 + 32, 64);
        v8f z = {};
        v8f acc0 = wmma_acc(a0, b0, z);
        v8f acc1 = wmma_acc(a1, b1, z);
        v8f acc = acc0 + acc1;
        int n = nt * 16 + (lane & 15);
        int mofs = (lane < 16) ? 0 : 8;
        for (int i = 0; i < 8; ++i)
            Ssc[(mt * 16 + i + mofs) * 116 + n] = acc[i];
    }
    __syncthreads();

    for (int r = wave; r < MPAD; r += 4) {
        if (r >= NTOK) {
            for (int j = 0; j < 4; ++j) P[r * PLD + lane + 32 * j] = (_Float16)0.f;
            continue;
        }
        int dn = r / 49, hn = (r % 49) / 7, wn = r % 7;
        float v[4];
        float mx = -3.0e38f;
        for (int j = 0; j < 4; ++j) {
            int cc = lane + 32 * j;
            if (cc < NTOK) {
                int dm = cc / 49, hm = (cc % 49) / 7, wm = cc % 7;
                int idx = (dn - dm + 1) * 169 + (hn - hm + 6) * 13 + (wn - wm + 6);
                v[j] = Ssc[r * 116 + cc] + bt[idx * 4 + head];
            } else {
                v[j] = -3.0e38f;
            }
            mx = fmaxf(mx, v[j]);
        }
        for (int off = 16; off; off >>= 1) mx = fmaxf(mx, __shfl_xor(mx, off, 32));
        float s = 0.f, e[4];
        for (int j = 0; j < 4; ++j) {
            int cc = lane + 32 * j;
            e[j] = (cc < NTOK) ? __expf(v[j] - mx) : 0.f;
            s += e[j];
        }
        for (int off = 16; off; off >>= 1) s += __shfl_xor(s, off, 32);
        float inv = 1.f / s;
        for (int j = 0; j < 4; ++j) {
            int cc = lane + 32 * j; // covers K range 0..127 for PV
            P[r * PLD + cc] = (_Float16)(e[j] * inv);
        }
    }
    __syncthreads();

    for (int tile = wave; tile < 28; tile += 4) {
        int mt = tile >> 2, nt = tile & 3;
        v8f acc0 = {}, acc1 = {};
        for (int kk = 0; kk < 4; kk += 2) {
            v16h a0 = load_frag_a(&P[mt * 16 * PLD + kk * 32], PLD);
            v16h b0 = load_frag_b(vg + (size_t)(nt * 16) * 128 + kk * 32, 128);
            acc0 = wmma_acc(a0, b0, acc0);
            v16h a1 = load_frag_a(&P[mt * 16 * PLD + (kk + 1) * 32], PLD);
            v16h b1 = load_frag_b(vg + (size_t)(nt * 16) * 128 + (kk + 1) * 32, 128);
            acc1 = wmma_acc(a1, b1, acc1);
        }
        v8f acc = acc0 + acc1;
        int n = nt * 16 + (lane & 15);
        int mofs = (lane < 16) ? 0 : 8;
        for (int i = 0; i < 8; ++i) {
            int m = mt * 16 + i + mofs;
            AO[((size_t)wi * 112 + m) * 256 + head * 64 + n] = (_Float16)acc[i];
        }
    }
}

// ---------------------------------------------------------------------------
// Output projection (112x256)x(256x256)+bias, scatter to voxel layout.
// gridMode==0: XW[vox][c] = proj            (window attention output)
// gridMode==1: S[vox][c]  = proj + XW + input  (final pre-conv activations)
// ---------------------------------------------------------------------------
__global__ __launch_bounds__(256)
void proj_kernel(const _Float16* __restrict__ AO,
                 const _Float16* __restrict__ Wh,
                 const float* __restrict__ pb, int gridMode,
                 const float* __restrict__ inp,
                 float* __restrict__ XW,
                 float* __restrict__ S) {
    __shared__ _Float16 As[MPAD * 264];
    int wi = blockIdx.x;
    int b = wi >> 8;
    int rem = wi & 255;
    int gd = rem >> 6, gh = (rem >> 3) & 7, gwn = rem & 7;
    int tid = threadIdx.x;

    const _Float16* Ag = AO + (size_t)wi * MPAD * 256;
    for (int idx = tid; idx < MPAD * 32; idx += 256) {
        int row = idx >> 5, cv = idx & 31;
        *(uint4*)(&As[row * 264 + cv * 8]) = *(const uint4*)(Ag + row * 256 + cv * 8);
    }
    __syncthreads();

    int wave = tid >> 5, lane = tid & 31;
    for (int tile = wave; tile < 7 * 16; tile += 8) {
        int mt = tile >> 4, nt = tile & 15;
        const _Float16* wrow = Wh + (size_t)(nt * 16) * 256;
        __builtin_prefetch(wrow, 0, 1);
        v8f acc0 = {}, acc1 = {};
        for (int kk = 0; kk < 8; kk += 2) {
            v16h a0 = load_frag_a(&As[mt * 16 * 264 + kk * 32], 264);
            v16h b0 = load_frag_b(wrow + kk * 32, 256);
            acc0 = wmma_acc(a0, b0, acc0);
            v16h a1 = load_frag_a(&As[mt * 16 * 264 + (kk + 1) * 32], 264);
            v16h b1 = load_frag_b(wrow + (kk + 1) * 32, 256);
            acc1 = wmma_acc(a1, b1, acc1);
        }
        v8f acc = acc0 + acc1;
        int n = nt * 16 + (lane & 15);
        float bv = pb[n];
        int mofs = (lane < 16) ? 0 : 8;
        for (int i = 0; i < 8; ++i) {
            int t = mt * 16 + i + mofs;
            if (t >= NTOK) continue;
            int dz = t / 49, hy = (t % 49) / 7, wx = t % 7;
            int dv, hv, wv;
            if (gridMode) { dv = dz * 4 + gd; hv = hy * 8 + gh; wv = wx * 8 + gwn; }
            else          { dv = gd * 2 + dz; hv = gh * 7 + hy; wv = gwn * 7 + wx; }
            size_t vox = ((size_t)(b * 8 + dv) * 56 + hv) * 56 + wv;
            float val = acc[i] + bv;
            if (!gridMode) {
                XW[vox * 256 + n] = val;
            } else {
                float r = inp[((size_t)(b * 256 + n) * 8 + dv) * 3136 + hv * 56 + wv];
                S[vox * 256 + n] = val + XW[vox * 256 + n] + r;
            }
        }
    }
}

// S channel-last upper half -> f16 conv input
__global__ void extract_x2_kernel(const float* __restrict__ S,
                                  _Float16* __restrict__ X2h) {
    int idx = blockIdx.x * blockDim.x + threadIdx.x;
    if (idx < NPOS * 128) {
        int p = idx >> 7, c = idx & 127;
        X2h[idx] = (_Float16)S[(size_t)p * 256 + 128 + c];
    }
}

// ---------------------------------------------------------------------------
// 3x3x3 conv 128->128, SAME, as 27 tap-wise WMMA GEMMs.
// One workgroup per (b,d,h) row: M=56 (pad 64) positions x N=128 oc,
// K = 128 ic per tap. finalMode==0: leaky -> f16 out. finalMode==1:
// + bias + residual (S channel-last, resOff), store f32 channel-first to
// d_out (+ optional f16 copy for the next conv).
// ---------------------------------------------------------------------------
__global__ __launch_bounds__(256)
void conv3d_kernel(const _Float16* __restrict__ Xin,
                   const _Float16* __restrict__ Wt,
                   const float* __restrict__ bias, int finalMode,
                   const float* __restrict__ Sres, int resOff,
                   float* __restrict__ outF, int outOff,
                   _Float16* __restrict__ outH) {
    int bid = blockIdx.x;
    int b = bid / (8 * 56);
    int rem = bid % (8 * 56);
    int d = rem / 56;
    int h = rem % 56;
    int wave = threadIdx.x >> 5, lane = threadIdx.x & 31;

    for (int tile = wave; tile < 32; tile += 8) {
        int mt = tile >> 3, nt = tile & 7;
        v8f acc0 = {}, acc1 = {};
        for (int tap = 0; tap < 27; ++tap) {
            int kd = tap / 9 - 1;
            int kh = (tap / 3) % 3 - 1;
            int kw = tap % 3 - 1;
            int dd = d + kd, hh = h + kh;
            if (dd < 0 || dd >= 8 || hh < 0 || hh >= 56) continue;
            const _Float16* row0 = Xin + ((size_t)((b * 8 + dd) * 56 + hh) * 56) * 128;
            const _Float16* wt = Wt + (size_t)tap * 16384 + (size_t)(nt * 16) * 128;
            if (tap < 26) __builtin_prefetch(wt + 16384, 0, 1); // next tap's weights
            int wbase = mt * 16 + kw;
            for (int kk = 0; kk < 4; kk += 2) {
                v16h a0 = load_frag_a_conv(row0, wbase, kk * 32);
                v16h b0 = load_frag_b(wt + kk * 32, 128);
                acc0 = wmma_acc(a0, b0, acc0);
                v16h a1 = load_frag_a_conv(row0, wbase, (kk + 1) * 32);
                v16h b1 = load_frag_b(wt + (kk + 1) * 32, 128);
                acc1 = wmma_acc(a1, b1, acc1);
            }
        }
        v8f acc = acc0 + acc1;
        int n = nt * 16 + (lane & 15);
        float bv = bias[n];
        int mofs = (lane < 16) ? 0 : 8;
        for (int i = 0; i < 8; ++i) {
            int m = mt * 16 + i + mofs;
            if (m >= 56) continue;
            size_t pos = (size_t)((b * 8 + d) * 56 + h) * 56 + m;
            float v = acc[i] + bv;
            if (!finalMode) {
                v = (v >= 0.f) ? v : 0.01f * v;
                outH[pos * 128 + n] = (_Float16)v;
            } else {
                v += Sres[pos * 256 + resOff + n];
                outF[((size_t)(b * 256 + outOff + n) * 8 + d) * 3136 + h * 56 + m] = v;
                if (outH) outH[pos * 128 + n] = (_Float16)v;
            }
        }
    }
}

// ---------------------------------------------------------------------------
extern "C" void kernel_launch(void* const* d_in, const int* in_sizes, int n_in,
                              void* d_out, int out_size, void* d_ws, size_t ws_size,
                              hipStream_t stream) {
    const float* input  = (const float*)d_in[0];
    const float* n1w    = (const float*)d_in[1];
    const float* n1b    = (const float*)d_in[2];
    const float* n2w    = (const float*)d_in[3];
    const float* n2b    = (const float*)d_in[4];
    const float* wqkv   = (const float*)d_in[5];
    const float* wprojw = (const float*)d_in[6];
    const float* wprojb = (const float*)d_in[7];
    const float* wbias  = (const float*)d_in[8];
    const float* gqkv   = (const float*)d_in[9];
    const float* gprojw = (const float*)d_in[10];
    const float* gprojb = (const float*)d_in[11];
    const float* gbias  = (const float*)d_in[12];
    const float* f1c1w  = (const float*)d_in[13];
    const float* f1c1b  = (const float*)d_in[14];
    const float* f1c2w  = (const float*)d_in[15];
    const float* f1c2b  = (const float*)d_in[16];
    const float* g1c1w  = (const float*)d_in[17];
    const float* g1c1b  = (const float*)d_in[18];
    const float* g1c2w  = (const float*)d_in[19];
    const float* g1c2b  = (const float*)d_in[20];
    float* out = (float*)d_out;

    char* ws = (char*)d_ws;
    size_t off = 0;
    auto alloc = [&](size_t bytes) -> char* {
        off = (off + 255) & ~size_t(255);
        char* p = ws + off;
        off += bytes;
        return p;
    };

    _Float16* wqkvH  = (_Float16*)alloc((size_t)768 * 256 * 2);
    _Float16* wprojH = (_Float16*)alloc((size_t)256 * 256 * 2);
    _Float16* gqkvH  = (_Float16*)alloc((size_t)768 * 256 * 2);
    _Float16* gprojH = (_Float16*)alloc((size_t)256 * 256 * 2);
    _Float16* Wc0 = (_Float16*)alloc((size_t)27 * 128 * 128 * 2);
    _Float16* Wc1 = (_Float16*)alloc((size_t)27 * 128 * 128 * 2);
    _Float16* Wc2 = (_Float16*)alloc((size_t)27 * 128 * 128 * 2);
    _Float16* Wc3 = (_Float16*)alloc((size_t)27 * 128 * 128 * 2);
    _Float16* Xtok = (_Float16*)alloc((size_t)NWIN * MPAD * 256 * 2);
    _Float16* qB   = (_Float16*)alloc((size_t)NWIN * 4 * 112 * 64 * 2);
    _Float16* kB   = (_Float16*)alloc((size_t)NWIN * 4 * 112 * 64 * 2);
    _Float16* vtB  = (_Float16*)alloc((size_t)NWIN * 4 * 64 * 128 * 2);
    _Float16* AO   = (_Float16*)alloc((size_t)NWIN * MPAD * 256 * 2);
    float*    XW   = (float*)alloc((size_t)NPOS * 256 * 4);
    float*    Sbuf = (float*)alloc((size_t)NPOS * 256 * 4);
    _Float16* X2h  = (_Float16*)alloc((size_t)NPOS * 128 * 2);
    _Float16* H1   = (_Float16*)alloc((size_t)NPOS * 128 * 2);
    _Float16* Y1h  = (_Float16*)alloc((size_t)NPOS * 128 * 2);
    _Float16* H2   = (_Float16*)alloc((size_t)NPOS * 128 * 2);

    // ---- weight conversion ----
    cvt_f16_kernel<<<(768 * 256 + 255) / 256, 256, 0, stream>>>(wqkv, wqkvH, 768 * 256);
    cvt_f16_kernel<<<(256 * 256 + 255) / 256, 256, 0, stream>>>(wprojw, wprojH, 256 * 256);
    cvt_f16_kernel<<<(768 * 256 + 255) / 256, 256, 0, stream>>>(gqkv, gqkvH, 768 * 256);
    cvt_f16_kernel<<<(256 * 256 + 255) / 256, 256, 0, stream>>>(gprojw, gprojH, 256 * 256);
    int nconvw = 27 * 128 * 128;
    cvt_convw_kernel<<<(nconvw + 255) / 256, 256, 0, stream>>>(f1c1w, Wc0);
    cvt_convw_kernel<<<(nconvw + 255) / 256, 256, 0, stream>>>(f1c2w, Wc1);
    cvt_convw_kernel<<<(nconvw + 255) / 256, 256, 0, stream>>>(g1c1w, Wc2);
    cvt_convw_kernel<<<(nconvw + 255) / 256, 256, 0, stream>>>(g1c2w, Wc3);

    // ---- window attention ----
    ln_gather_kernel<<<NWIN, 256, 0, stream>>>(input, n1w, n1b, Xtok, 0);
    qkv_gemm_kernel<<<NWIN, 256, 0, stream>>>(Xtok, wqkvH, qB, kB, vtB);
    attn_core_kernel<<<NWIN * 4, 128, 0, stream>>>(qB, kB, vtB, wbias, AO);
    proj_kernel<<<NWIN, 256, 0, stream>>>(AO, wprojH, wprojb, 0, nullptr, XW, nullptr);

    // ---- grid attention (reads XW, fuses +XW+input into S) ----
    ln_gather_kernel<<<NWIN, 256, 0, stream>>>(XW, n2w, n2b, Xtok, 1);
    qkv_gemm_kernel<<<NWIN, 256, 0, stream>>>(Xtok, gqkvH, qB, kB, vtB);
    attn_core_kernel<<<NWIN * 4, 128, 0, stream>>>(qB, kB, vtB, gbias, AO);
    proj_kernel<<<NWIN, 256, 0, stream>>>(AO, gprojH, gprojb, 1, input, XW, Sbuf);

    // ---- conv FFN halves ----
    extract_x2_kernel<<<(NPOS * 128 + 255) / 256, 256, 0, stream>>>(Sbuf, X2h);
    int nrow = 2 * 8 * 56;
    conv3d_kernel<<<nrow, 256, 0, stream>>>(X2h, Wc0, f1c1b, 0, nullptr, 0, nullptr, 0, H1);
    conv3d_kernel<<<nrow, 256, 0, stream>>>(H1, Wc1, f1c2b, 1, Sbuf, 0, out, 0, Y1h);
    conv3d_kernel<<<nrow, 256, 0, stream>>>(Y1h, Wc2, g1c1b, 0, nullptr, 0, nullptr, 0, H2);
    conv3d_kernel<<<nrow, 256, 0, stream>>>(H2, Wc3, g1c2b, 1, Sbuf, 128, out, 128, nullptr);

    (void)in_sizes; (void)n_in; (void)out_size; (void)ws_size;
}
